// EdgeEncoder_85521388798454
// MI455X (gfx1250) — matrix operand order repaired and checked by reading, not compiled, not run on hardware
//
#include <hip/hip_runtime.h>
#include <hip/hip_bf16.h>
#include <stdint.h>

typedef __attribute__((ext_vector_type(16))) _Float16 v16h;
typedef __attribute__((ext_vector_type(8)))  _Float16 v8h;
typedef __attribute__((ext_vector_type(8)))  float    v8f;

#define LN_EPS 1e-5f

// ---------------------------------------------------------------------------
// WMMA fragment loaders (gfx1250 16x16x32 f16 layouts, CDNA5 ISA)
// ---------------------------------------------------------------------------
// A fragment: row-major A in LDS, caller passes pointer to this lane's row.
// lanes 0-15 : K = {k0+0..7, k0+16..23}; lanes 16-31: K = {k0+8..15, k0+24..31}
__device__ __forceinline__ v16h load_a_frag(const _Float16* row, int k0, int lane) {
  const int sel = (lane & 16) ? 8 : 0;
  v16h r;
  ((v8h*)&r)[0] = *(const v8h*)(row + k0 + sel);
  ((v8h*)&r)[1] = *(const v8h*)(row + k0 + sel + 16);
  return r;
}
// B fragment from pre-swizzled weights: Wp laid out [ntile][kstep][lane][16h],
// so each wave fragment fetch is one contiguous, fully coalesced 1KB read.
__device__ __forceinline__ v16h load_b_packed(const _Float16* Wp, int frag, int lane) {
  return *(const v16h*)(Wp + (((size_t)frag * 32 + lane) << 4));
}

__device__ __forceinline__ v8f zero8() {
  v8f z;
#pragma unroll
  for (int i = 0; i < 8; ++i) z[i] = 0.0f;
  return z;
}

__device__ __forceinline__ void red4(float& a, float& b) {
  a += __shfl_xor(a, 1, 32);  b += __shfl_xor(b, 1, 32);
  a += __shfl_xor(a, 2, 32);  b += __shfl_xor(b, 2, 32);
}

// monotonic float->uint key so segment-max can use u32 atomicMax; 0 == empty
__device__ __forceinline__ unsigned fkey(float v) {
  unsigned b = __float_as_uint(v);
  return (b & 0x80000000u) ? ~b : (b | 0x80000000u);
}
__device__ __forceinline__ float fkey_dec(unsigned k) {
  if (k == 0u) return 0.0f;                       // empty segment -> 0 (ref -inf->0)
  unsigned b = (k & 0x80000000u) ? (k & 0x7FFFFFFFu) : ~k;
  return __uint_as_float(b);
}

// ---------------------------------------------------------------------------
// Weight prep: fp32 W[KxN] -> f16 packed fragment order
// Wp[((nt*KS + ks)*32 + lane)*16 + h] = W[(ks*32 + (lane&16 ? 8:0) + (h&7) + (h&8 ? 16:0)) * N
//                                        + nt*16 + (lane&15)]
// ---------------------------------------------------------------------------
__global__ void wt_pack_kernel(const float* __restrict__ W,
                               _Float16* __restrict__ Wp, int K, int N) {
  int idx = blockIdx.x * blockDim.x + threadIdx.x;
  if (idx >= K * N) return;
  const int KS = K >> 5;
  int h    = idx & 15;
  int lane = (idx >> 4) & 31;
  int rem  = idx >> 9;
  int ks   = rem % KS;
  int nt   = rem / KS;
  int n = nt * 16 + (lane & 15);
  int k = ks * 32 + ((lane & 16) ? 8 : 0) + (h & 7) + ((h & 8) ? 16 : 0);
  Wp[idx] = (_Float16)W[(size_t)k * N + n];
}

// ---------------------------------------------------------------------------
// Kernel A: gather/embed + 3-layer MLP (64->128->256->256) + scatter-max
// 256 threads (8 waves), TM = 128 edges/block, ~118 KB LDS.
// Wave w owns N-tile column(s); per k-step: 1 B-frag feeds 8 WMMAs (all M-tiles)
// => each weight matrix is read exactly once per block.
// ---------------------------------------------------------------------------
__global__ __launch_bounds__(256) void edge_mlp_kernel(
    const int* __restrict__ rel_type, const float* __restrict__ rel_error,
    const int* __restrict__ seg_row,  const int* __restrict__ rel_pos,
    const float* __restrict__ type_emb, const float* __restrict__ pos_emb,
    const _Float16* __restrict__ W1p, const float* __restrict__ b1,
    const _Float16* __restrict__ W2p, const float* __restrict__ b2,
    const _Float16* __restrict__ W3p, const float* __restrict__ b3,
    unsigned* __restrict__ segKeys, int E)
{
  __shared__ _Float16 Xs [128][72];    // 18 KB, 144B stride
  __shared__ _Float16 H1s[128][136];   // 34 KB, 272B stride
  __shared__ _Float16 H2s[128][264];   // 66 KB, 528B stride

  const int tid  = threadIdx.x;
  const int lane = tid & 31;
  const int wave = tid >> 5;
  const int ln   = lane & 15;
  const int mloc = (lane >> 4) * 8;     // C/D row offset for this lane half
  const int e0   = blockIdx.x * 128;

  // ---- stage 1: X = concat(type_emb[t], rel_error) + pos_emb[p], fp32 -> f16
  for (int i = tid; i < 128 * 64; i += 256) {
    int r = i >> 6, c = i & 63;
    int e = e0 + r;
    float v = 0.0f;
    if (e < E) {
      int t = rel_type[e];
      int p = rel_pos[e];
      float base = (c < 56) ? type_emb[t * 56 + c] : rel_error[(size_t)e * 8 + (c - 56)];
      v = base + pos_emb[p * 64 + c];
    }
    Xs[r][c] = (_Float16)v;
  }
  __syncthreads();

  // ---- stage 2: H1 = relu(X @ W1 + b1)  M=128 N=128 K=64; wave w owns nt=w
  {
    v8f acc[8];
#pragma unroll
    for (int m = 0; m < 8; ++m) acc[m] = zero8();
#pragma unroll
    for (int ks = 0; ks < 2; ++ks) {
      v16h b = load_b_packed(W1p, wave * 2 + ks, lane);
#pragma unroll
      for (int m = 0; m < 8; ++m) {
        v16h a = load_a_frag(&Xs[m * 16 + ln][0], ks * 32, lane);
        acc[m] = __builtin_amdgcn_wmma_f32_16x16x32_f16(false, a, false, b,
                                                        (short)0, acc[m], false, false);
      }
    }
    const int nb = wave * 16 + ln;
    const float bias = b1[nb];
#pragma unroll
    for (int m = 0; m < 8; ++m)
#pragma unroll
      for (int r = 0; r < 8; ++r) {
        float v = acc[m][r] + bias;
        H1s[m * 16 + mloc + r][nb] = (_Float16)(v > 0.0f ? v : 0.0f);
      }
  }
  __syncthreads();

  // ---- stage 3: H2 = relu(H1 @ W2 + b2)  M=128 N=256 K=128; nt = wave + 8p
  for (int p = 0; p < 2; ++p) {
    const int nt = wave + p * 8;
    v8f acc[8];
#pragma unroll
    for (int m = 0; m < 8; ++m) acc[m] = zero8();
#pragma unroll
    for (int ks = 0; ks < 4; ++ks) {
      v16h b = load_b_packed(W2p, nt * 4 + ks, lane);
#pragma unroll
      for (int m = 0; m < 8; ++m) {
        v16h a = load_a_frag(&H1s[m * 16 + ln][0], ks * 32, lane);
        acc[m] = __builtin_amdgcn_wmma_f32_16x16x32_f16(false, a, false, b,
                                                        (short)0, acc[m], false, false);
      }
    }
    const int nb = nt * 16 + ln;
    const float bias = b2[nb];
#pragma unroll
    for (int m = 0; m < 8; ++m)
#pragma unroll
      for (int r = 0; r < 8; ++r) {
        float v = acc[m][r] + bias;
        H2s[m * 16 + mloc + r][nb] = (_Float16)(v > 0.0f ? v : 0.0f);
      }
  }
  __syncthreads();

  // ---- stage 4: H3 = H2 @ W3 + b3, monotonic-key atomic scatter-max
  for (int p = 0; p < 2; ++p) {
    const int nt = wave + p * 8;
    if (p == 0) {  // prefetch next pass's 8KB weight slice (global_prefetch_b8)
      const char* nxt = (const char*)(W3p + ((size_t)(wave + 8) * 8) * 512);
      __builtin_prefetch(nxt + lane * 256, 0, 1);
    }
    v8f acc[8];
#pragma unroll
    for (int m = 0; m < 8; ++m) acc[m] = zero8();
#pragma unroll
    for (int ks = 0; ks < 8; ++ks) {
      v16h b = load_b_packed(W3p, nt * 8 + ks, lane);
#pragma unroll
      for (int m = 0; m < 8; ++m) {
        v16h a = load_a_frag(&H2s[m * 16 + ln][0], ks * 32, lane);
        acc[m] = __builtin_amdgcn_wmma_f32_16x16x32_f16(false, a, false, b,
                                                        (short)0, acc[m], false, false);
      }
    }
    const int nb = nt * 16 + ln;
    const float bias = b3[nb];
#pragma unroll
    for (int m = 0; m < 8; ++m)
#pragma unroll
      for (int r = 0; r < 8; ++r) {
        int e = e0 + m * 16 + mloc + r;
        if (e < E) {
          unsigned key = fkey(acc[m][r] + bias);
          int s = seg_row[e];
          atomicMax(&segKeys[(size_t)s * 256 + nb], key);
        }
      }
  }
}

// ---------------------------------------------------------------------------
// Kernel B: seg decode + dist_emb + LN -> GEMM(256x256)+relu -> LN ->
//           GEMM(256x32) -> LN -> out.  256 threads, TM = 64 rows, ~98 KB LDS.
// ---------------------------------------------------------------------------
__global__ __launch_bounds__(256) void fc_kernel(
    const unsigned* __restrict__ segKeys, const int* __restrict__ dist,
    const float* __restrict__ dist_emb,
    const float* __restrict__ g1, const float* __restrict__ be1,
    const _Float16* __restrict__ W4p, const float* __restrict__ b4,
    const float* __restrict__ g2, const float* __restrict__ be2,
    const _Float16* __restrict__ W5p, const float* __restrict__ b5,
    const float* __restrict__ g3, const float* __restrict__ be3,
    float* __restrict__ out, int num_seg)
{
  __shared__ float    S32[64][260];   // 65 KB fp32 staging (pre-LN values / output)
  __shared__ _Float16 A16[64][264];   // 33 KB f16 normalized A operand

  const int tid  = threadIdx.x;
  const int lane = tid & 31;
  const int wave = tid >> 5;
  const int ln   = lane & 15;
  const int mloc = (lane >> 4) * 8;
  const int s0   = blockIdx.x * 64;

  // ---- stage 0: decode segment max + dist_emb lookup
  for (int i = tid; i < 64 * 256; i += 256) {
    int r = i >> 8, c = i & 255;
    int s = s0 + r;
    float v = 0.0f;
    if (s < num_seg) {
      v = fkey_dec(segKeys[(size_t)s * 256 + c]) + dist_emb[dist[s] * 256 + c];
    }
    S32[r][c] = v;
  }
  __syncthreads();

  // ---- LN1 (256 dims, 4 lanes per row) -> A16 (f16)
  {
    const int r = tid >> 2, q = tid & 3;
    float sum = 0.0f, ssq = 0.0f;
#pragma unroll 4
    for (int c = q * 64; c < q * 64 + 64; ++c) { float v = S32[r][c]; sum += v; ssq += v * v; }
    red4(sum, ssq);
    float mu = sum * (1.0f / 256.0f);
    float rs = rsqrtf(ssq * (1.0f / 256.0f) - mu * mu + LN_EPS);
#pragma unroll 4
    for (int c = q * 64; c < q * 64 + 64; ++c)
      A16[r][c] = (_Float16)((S32[r][c] - mu) * rs * g1[c] + be1[c]);
  }
  __syncthreads();

  // ---- GEMM1: relu(A @ W4 + b4)  M=64 N=256 K=256; wave w owns nt = w + 8p
  for (int p = 0; p < 2; ++p) {
    const int nt = wave + p * 8;
    if (p == 0) {
      const char* nxt = (const char*)(W4p + ((size_t)(wave + 8) * 8) * 512);
      __builtin_prefetch(nxt + lane * 256, 0, 1);
    }
    v8f acc[4];
#pragma unroll
    for (int m = 0; m < 4; ++m) acc[m] = zero8();
#pragma unroll
    for (int ks = 0; ks < 8; ++ks) {
      v16h b = load_b_packed(W4p, nt * 8 + ks, lane);
#pragma unroll
      for (int m = 0; m < 4; ++m) {
        v16h a = load_a_frag(&A16[m * 16 + ln][0], ks * 32, lane);
        acc[m] = __builtin_amdgcn_wmma_f32_16x16x32_f16(false, a, false, b,
                                                        (short)0, acc[m], false, false);
      }
    }
    const int nb = nt * 16 + ln;
    const float bias = b4[nb];
#pragma unroll
    for (int m = 0; m < 4; ++m)
#pragma unroll
      for (int r = 0; r < 8; ++r) {
        float v = acc[m][r] + bias;
        S32[m * 16 + mloc + r][nb] = v > 0.0f ? v : 0.0f;
      }
  }
  __syncthreads();

  // ---- LN2 -> A16
  {
    const int r = tid >> 2, q = tid & 3;
    float sum = 0.0f, ssq = 0.0f;
#pragma unroll 4
    for (int c = q * 64; c < q * 64 + 64; ++c) { float v = S32[r][c]; sum += v; ssq += v * v; }
    red4(sum, ssq);
    float mu = sum * (1.0f / 256.0f);
    float rs = rsqrtf(ssq * (1.0f / 256.0f) - mu * mu + LN_EPS);
#pragma unroll 4
    for (int c = q * 64; c < q * 64 + 64; ++c)
      A16[r][c] = (_Float16)((S32[r][c] - mu) * rs * g2[c] + be2[c]);
  }
  __syncthreads();

  // ---- GEMM2: A @ W5 + b5   M=64 N=32 K=256 (4x2 tiles, 1/wave) -> S32[:, :32]
  {
    v8f acc = zero8();
    const int mt = wave >> 1, nt = wave & 1;
#pragma unroll
    for (int ks = 0; ks < 8; ++ks) {
      v16h a = load_a_frag(&A16[mt * 16 + ln][0], ks * 32, lane);
      v16h b = load_b_packed(W5p, nt * 8 + ks, lane);
      acc = __builtin_amdgcn_wmma_f32_16x16x32_f16(false, a, false, b,
                                                   (short)0, acc, false, false);
    }
    int nb = nt * 16 + ln;
    float bias = b5[nb];
#pragma unroll
    for (int r = 0; r < 8; ++r)
      S32[mt * 16 + mloc + r][nb] = acc[r] + bias;
  }
  __syncthreads();

  // ---- LN3 (32 dims, 4 lanes per row) -> out
  {
    const int r = tid >> 2, q = tid & 3;
    float sum = 0.0f, ssq = 0.0f;
#pragma unroll
    for (int c = q * 8; c < q * 8 + 8; ++c) { float v = S32[r][c]; sum += v; ssq += v * v; }
    red4(sum, ssq);
    float mu = sum * (1.0f / 32.0f);
    float rs = rsqrtf(ssq * (1.0f / 32.0f) - mu * mu + LN_EPS);
    int s = s0 + r;
    if (s < num_seg) {
#pragma unroll
      for (int c = q * 8; c < q * 8 + 8; ++c)
        out[(size_t)s * 32 + c] = (S32[r][c] - mu) * rs * g3[c] + be3[c];
    }
  }
}

// ---------------------------------------------------------------------------
// launch
// ---------------------------------------------------------------------------
extern "C" void kernel_launch(void* const* d_in, const int* in_sizes, int n_in,
                              void* d_out, int out_size, void* d_ws, size_t ws_size,
                              hipStream_t stream) {
  const int*   rel_type  = (const int*)  d_in[0];
  const float* rel_error = (const float*)d_in[1];
  const int*   rel_coor  = (const int*)  d_in[2];   // [2,E]; row = first E
  const int*   rel_pos   = (const int*)  d_in[5];
  const int*   dist      = (const int*)  d_in[6];   // flat [B*N*N]
  const float* type_emb  = (const float*)d_in[7];
  const float* pos_emb   = (const float*)d_in[8];
  const float* dist_emb  = (const float*)d_in[9];
  const float* W1 = (const float*)d_in[10]; const float* b1  = (const float*)d_in[11];
  const float* W2 = (const float*)d_in[12]; const float* b2  = (const float*)d_in[13];
  const float* W3 = (const float*)d_in[14]; const float* b3  = (const float*)d_in[15];
  const float* g1 = (const float*)d_in[16]; const float* be1 = (const float*)d_in[17];
  const float* W4 = (const float*)d_in[18]; const float* b4  = (const float*)d_in[19];
  const float* g2 = (const float*)d_in[20]; const float* be2 = (const float*)d_in[21];
  const float* W5 = (const float*)d_in[22]; const float* b5  = (const float*)d_in[23];
  const float* g3 = (const float*)d_in[24]; const float* be3 = (const float*)d_in[25];

  const int E = in_sizes[0];
  const int num_seg = out_size / 32;       // D_REL = 32

  // workspace: [segment key buffer (u32, num_seg*256)] [f16 packed weights]
  uint8_t* ws = (uint8_t*)d_ws;
  unsigned* segKeys = (unsigned*)ws;
  size_t off = (size_t)num_seg * 256 * sizeof(unsigned);
  _Float16* W1p = (_Float16*)(ws + off); off += (size_t)128 * 64  * sizeof(_Float16);
  _Float16* W2p = (_Float16*)(ws + off); off += (size_t)256 * 128 * sizeof(_Float16);
  _Float16* W3p = (_Float16*)(ws + off); off += (size_t)256 * 256 * sizeof(_Float16);
  _Float16* W4p = (_Float16*)(ws + off); off += (size_t)256 * 256 * sizeof(_Float16);
  _Float16* W5p = (_Float16*)(ws + off); off += (size_t)32  * 256 * sizeof(_Float16);

  hipMemsetAsync(segKeys, 0, (size_t)num_seg * 256 * sizeof(unsigned), stream);

  wt_pack_kernel<<<(64  * 128 + 255) / 256, 256, 0, stream>>>(W1, W1p, 64,  128);
  wt_pack_kernel<<<(128 * 256 + 255) / 256, 256, 0, stream>>>(W2, W2p, 128, 256);
  wt_pack_kernel<<<(256 * 256 + 255) / 256, 256, 0, stream>>>(W3, W3p, 256, 256);
  wt_pack_kernel<<<(256 * 256 + 255) / 256, 256, 0, stream>>>(W4, W4p, 256, 256);
  wt_pack_kernel<<<(256 * 32  + 255) / 256, 256, 0, stream>>>(W5, W5p, 256, 32);

  edge_mlp_kernel<<<(E + 127) / 128, 256, 0, stream>>>(
      rel_type, rel_error, rel_coor, rel_pos, type_emb, pos_emb,
      W1p, b1, W2p, b2, W3p, b3, segKeys, E);

  fc_kernel<<<(num_seg + 63) / 64, 256, 0, stream>>>(
      segKeys, dist, dist_emb, g1, be1, W4p, b4, g2, be2, W5p, b5, g3, be3,
      (float*)d_out, num_seg);
}
